// DeepGAT_12017318494742
// MI455X (gfx1250) — matrix operand order repaired and checked by reading, not compiled, not run on hardware
//
#include <hip/hip_runtime.h>
#include <math.h>

#define N_NODES 10000
#define IN_DIM  512
#define HID     128
#define HEADS   8
#define LAYERS  4
#define NCLS    2
#define RES_ALPHA 0.1f
#define BN_EPS  1e-5f
#define NEG_SLOPE 0.2f

typedef __attribute__((ext_vector_type(16))) __bf16 v16bf;
typedef __attribute__((ext_vector_type(8)))  float  v8f;

union Frag32 {           // 32-byte WMMA operand fragment = two 16B LDS loads
  uint4 q[2];
  v16bf v;
};
static_assert(sizeof(v16bf) == 32, "v16bf must be 32 bytes");

__device__ __forceinline__ float elu_f(float x) { return x > 0.f ? x : (expf(x) - 1.f); }

// monotonic float <-> uint mapping for atomicMax on floats
__device__ __forceinline__ unsigned f2mono(float f) {
  unsigned u = __float_as_uint(f);
  return (u & 0x80000000u) ? ~u : (u | 0x80000000u);
}
__device__ __forceinline__ float mono2f(unsigned m) {
  unsigned u = (m & 0x80000000u) ? (m & 0x7FFFFFFFu) : ~m;
  return __uint_as_float(u);
}

// ---------------------------------------------------------------------------
// LDS-staged WMMA GEMM: D[M,Nc] = act(A[M,K] @ B[K,Nc] + bias)
// Block = 256 threads (8 waves) on a 128x64 output tile.
// Per K-step(32): coalesced float4 global loads -> bf16 in LDS (B transposed),
// then each wave reads its fragments as contiguous ds_load_b128 pairs and
// issues 4 v_wmma_f32_16x16x32_bf16 (one 16x64 strip per wave).
// Requires Nc%64==0, K%32==0. act: 0=none, 1=ELU.
// ---------------------------------------------------------------------------
#define AST 40   // A LDS row stride in bf16 units (80B: bank spread, 16B aligned)
#define BST 40   // Bt LDS row stride in bf16 units

__global__ void __launch_bounds__(256)
wmma_gemm_kernel(const float* __restrict__ A, const float* __restrict__ B,
                 const float* __restrict__ bias, float* __restrict__ D,
                 int M, int Nc, int K, int act) {
  __shared__ __bf16 A_sh[128 * AST];  // 10240 B
  __shared__ __bf16 B_sh[64 * BST];   //  5120 B (transposed: [col][k])

  const int groupsN = Nc >> 6;
  const int mb = blockIdx.x / groupsN;   // 128-row block
  const int tg = blockIdx.x % groupsN;   // 64-col group
  const int lane = threadIdx.x & 31;
  const int wave = threadIdx.x >> 5;     // 0..7 -> 16-row strip within block
  const int half = lane >> 4;
  const int r    = lane & 15;

  v8f acc0 = {}, acc1 = {}, acc2 = {}, acc3 = {};

  for (int k0 = 0; k0 < K; k0 += 32) {
    // ---- cooperative A panel load: 128x32 fp32 -> bf16 LDS (row-major) ----
    {
      const int colq  = threadIdx.x & 7;    // float4 index within 32-wide row
      const int rbase = threadIdx.x >> 3;   // 0..31
#pragma unroll
      for (int rep = 0; rep < 4; ++rep) {
        const int row = rep * 32 + rbase;
        int g = mb * 128 + row;
        if (g >= M) g = M - 1;  // clamp; out-of-range rows never stored
        const float4 f = *(const float4*)(A + (size_t)g * K + k0 + colq * 4);
        if (k0 + 32 < K) __builtin_prefetch(A + (size_t)g * K + k0 + 32 + colq * 4, 0, 0);
        __bf16* d = A_sh + row * AST + colq * 4;
        d[0] = (__bf16)f.x; d[1] = (__bf16)f.y; d[2] = (__bf16)f.z; d[3] = (__bf16)f.w;
      }
    }
    // ---- cooperative B panel load: 32x64 fp32 -> bf16 LDS transposed ----
    {
      const int k   = threadIdx.x >> 3;   // 0..31
      const int seg = threadIdx.x & 7;    // 8-col segment
      const float* brow = B + (size_t)(k0 + k) * Nc + tg * 64 + seg * 8;
      const float4 f0 = *(const float4*)brow;
      const float4 f1 = *(const float4*)(brow + 4);
      const int cb = seg * 8;
      B_sh[(cb + 0) * BST + k] = (__bf16)f0.x;
      B_sh[(cb + 1) * BST + k] = (__bf16)f0.y;
      B_sh[(cb + 2) * BST + k] = (__bf16)f0.z;
      B_sh[(cb + 3) * BST + k] = (__bf16)f0.w;
      B_sh[(cb + 4) * BST + k] = (__bf16)f1.x;
      B_sh[(cb + 5) * BST + k] = (__bf16)f1.y;
      B_sh[(cb + 6) * BST + k] = (__bf16)f1.z;
      B_sh[(cb + 7) * BST + k] = (__bf16)f1.w;
    }
    __syncthreads();

    // ---- fragments from LDS ----
    // A 16x32 layout: lanes 0-15 hold K 0..7,16..23 ; lanes 16-31 hold K 8..15,24..31
    Frag32 fa;
    {
      const __bf16* ap = A_sh + (wave * 16 + r) * AST + 8 * half;
      fa.q[0] = *(const uint4*)ap;         // K = 8*half + 0..7
      fa.q[1] = *(const uint4*)(ap + 16);  // K = 16 + 8*half + 0..7
    }
    // B 32x16 layout: lanes 0-15 hold K 0..7,16..23 ; lanes 16-31 hold K 8..15,24..31
    Frag32 fb0, fb1, fb2, fb3;
    {
      const __bf16* bp = B_sh + r * BST + 8 * half;
      fb0.q[0] = *(const uint4*)bp;        fb0.q[1] = *(const uint4*)(bp + 16);
      bp += 16 * BST;
      fb1.q[0] = *(const uint4*)bp;        fb1.q[1] = *(const uint4*)(bp + 16);
      bp += 16 * BST;
      fb2.q[0] = *(const uint4*)bp;        fb2.q[1] = *(const uint4*)(bp + 16);
      bp += 16 * BST;
      fb3.q[0] = *(const uint4*)bp;        fb3.q[1] = *(const uint4*)(bp + 16);
    }
    acc0 = __builtin_amdgcn_wmma_f32_16x16x32_bf16(false, fa.v, false, fb0.v, (short)0, acc0, false, false);
    acc1 = __builtin_amdgcn_wmma_f32_16x16x32_bf16(false, fa.v, false, fb1.v, (short)0, acc1, false, false);
    acc2 = __builtin_amdgcn_wmma_f32_16x16x32_bf16(false, fa.v, false, fb2.v, (short)0, acc2, false, false);
    acc3 = __builtin_amdgcn_wmma_f32_16x16x32_bf16(false, fa.v, false, fb3.v, (short)0, acc3, false, false);
    __syncthreads();
  }

  // C/D layout: VGPR j -> M = j + 8*half, N = lane%16 (+ tile column offset)
#pragma unroll
  for (int j = 0; j < 8; ++j) {
    const int m = mb * 128 + wave * 16 + j + 8 * half;
    if (m >= M) continue;
    const int nbase = tg * 64 + r;
    float v0 = acc0[j], v1 = acc1[j], v2 = acc2[j], v3 = acc3[j];
    if (bias) {
      v0 += bias[nbase];
      v1 += bias[nbase + 16];
      v2 += bias[nbase + 32];
      v3 += bias[nbase + 48];
    }
    if (act) { v0 = elu_f(v0); v1 = elu_f(v1); v2 = elu_f(v2); v3 = elu_f(v3); }
    float* drow = D + (size_t)m * Nc + nbase;
    drow[0]  = v0;
    drow[16] = v1;
    drow[32] = v2;
    drow[48] = v3;
  }
}

// ---------------------------------------------------------------------------
// alpha_s[n,h] = <xl[n,h,:], a_src[h,:]> ; alpha_d likewise
// ---------------------------------------------------------------------------
__global__ void alpha_kernel(const float* __restrict__ xl,
                             const float* __restrict__ a_src,
                             const float* __restrict__ a_dst,
                             float* __restrict__ alpha_s,
                             float* __restrict__ alpha_d) {
  int idx = blockIdx.x * blockDim.x + threadIdx.x;  // n*HEADS + h
  if (idx >= N_NODES * HEADS) return;
  int n = idx / HEADS, h = idx % HEADS;
  const float* xr = xl + (size_t)n * (HEADS * HID) + h * HID;
  const float* as = a_src + h * HID;
  const float* ad = a_dst + h * HID;
  float ss = 0.f, sd = 0.f;
  for (int c = 0; c < HID; c += 4) {
    float4 xv = *(const float4*)(xr + c);
    float4 sv = *(const float4*)(as + c);
    float4 dv = *(const float4*)(ad + c);
    ss += xv.x * sv.x + xv.y * sv.y + xv.z * sv.z + xv.w * sv.w;
    sd += xv.x * dv.x + xv.y * dv.y + xv.z * dv.z + xv.w * dv.w;
  }
  alpha_s[idx] = ss;
  alpha_d[idx] = sd;
}

__global__ void zero_kernel(float* __restrict__ p, int n) {
  int i = blockIdx.x * blockDim.x + threadIdx.x;
  if (i < n) p[i] = 0.f;
}

// pass 1: per-(dst,head) running max of leaky_relu(alpha_s[src]+alpha_d[dst])
__global__ void edge_max_kernel(const int* __restrict__ ei, int E, int EP,
                                const float* __restrict__ alpha_s,
                                const float* __restrict__ alpha_d,
                                unsigned* __restrict__ emax) {
  int idx = blockIdx.x * blockDim.x + threadIdx.x;  // e*HEADS + h
  if (idx >= EP * HEADS) return;
  int e = idx / HEADS, h = idx % HEADS;
  int src = (e < E) ? ei[e] : (e - E);
  int dst = (e < E) ? ei[E + e] : (e - E);
  float x = alpha_s[src * HEADS + h] + alpha_d[dst * HEADS + h];
  x = x > 0.f ? x : NEG_SLOPE * x;
  atomicMax(&emax[dst * HEADS + h], f2mono(x));
}

// pass 2: ee = exp(e - max); denom[dst,h] += ee
__global__ void edge_exp_kernel(const int* __restrict__ ei, int E, int EP,
                                const float* __restrict__ alpha_s,
                                const float* __restrict__ alpha_d,
                                const unsigned* __restrict__ emax,
                                float* __restrict__ ee,
                                float* __restrict__ denom) {
  int idx = blockIdx.x * blockDim.x + threadIdx.x;
  if (idx >= EP * HEADS) return;
  int e = idx / HEADS, h = idx % HEADS;
  int src = (e < E) ? ei[e] : (e - E);
  int dst = (e < E) ? ei[E + e] : (e - E);
  float x = alpha_s[src * HEADS + h] + alpha_d[dst * HEADS + h];
  x = x > 0.f ? x : NEG_SLOPE * x;
  float v = expf(x - mono2f(emax[dst * HEADS + h]));
  ee[idx] = v;
  unsafeAtomicAdd(&denom[dst * HEADS + h], v);
}

// pass 3: head-reduced message. acc[dst,c] += (1/H) * sum_h att[h] * xl[src,h,c]
// (mean over heads commutes with the segment sum -> 8x fewer atomics)
__global__ void __launch_bounds__(HID)
message_kernel(const int* __restrict__ ei, int E,
               const float* __restrict__ xl, const float* __restrict__ ee,
               const float* __restrict__ denom, float* __restrict__ acc) {
  int e = blockIdx.x;
  int src = (e < E) ? ei[e] : (e - E);
  int dst = (e < E) ? ei[E + e] : (e - E);
  __shared__ float w[HEADS];
  if (threadIdx.x < HEADS) {
    float d = denom[dst * HEADS + threadIdx.x];
    w[threadIdx.x] = ee[(size_t)e * HEADS + threadIdx.x] / (d + 1e-16f) * (1.0f / HEADS);
  }
  __syncthreads();
  int c = threadIdx.x;
  const float* xr = xl + (size_t)src * (HEADS * HID) + c;
  float s = 0.f;
#pragma unroll
  for (int h = 0; h < HEADS; ++h) s += w[h] * xr[h * HID];
  unsafeAtomicAdd(&acc[(size_t)dst * HID + c], s);
}

// BN(eval) + residual + ELU
__global__ void post_kernel(const float* __restrict__ acc, const float* __restrict__ hprev,
                            const float* __restrict__ bl, const float* __restrict__ gamma,
                            const float* __restrict__ beta, float* __restrict__ hout) {
  int idx = blockIdx.x * blockDim.x + threadIdx.x;
  if (idx >= N_NODES * HID) return;
  int c = idx % HID;
  const float bn_inv = rsqrtf(1.f + BN_EPS);
  float g = acc[idx] + bl[c];
  g = gamma[c] * (g * bn_inv) + beta[c];
  float v = (1.f - RES_ALPHA) * g + RES_ALPHA * hprev[idx];
  hout[idx] = elu_f(v);
}

// final tiny classifier: out[n,c] = hid[n,:] @ W2[:,c] + b2[c]
__global__ void out_kernel(const float* __restrict__ hid, const float* __restrict__ W2,
                           const float* __restrict__ b2, float* __restrict__ out) {
  int idx = blockIdx.x * blockDim.x + threadIdx.x;
  if (idx >= N_NODES * NCLS) return;
  int n = idx / NCLS, c2 = idx % NCLS;
  float s = b2[c2];
  for (int k = 0; k < HID / 2; ++k) s += hid[(size_t)n * (HID / 2) + k] * W2[k * NCLS + c2];
  out[idx] = s;
}

extern "C" void kernel_launch(void* const* d_in, const int* in_sizes, int n_in,
                              void* d_out, int out_size, void* d_ws, size_t ws_size,
                              hipStream_t stream) {
  const float* x       = (const float*)d_in[0];
  const int*   ei      = (const int*)d_in[1];
  const float* Wp      = (const float*)d_in[2];
  const float* bp      = (const float*)d_in[3];
  const float* Wl      = (const float*)d_in[4];
  const float* att_src = (const float*)d_in[5];
  const float* att_dst = (const float*)d_in[6];
  const float* bl      = (const float*)d_in[7];
  const float* gamma   = (const float*)d_in[8];
  const float* beta    = (const float*)d_in[9];
  const float* W1      = (const float*)d_in[10];
  const float* b1      = (const float*)d_in[11];
  const float* W2      = (const float*)d_in[12];
  const float* b2      = (const float*)d_in[13];

  const int E  = in_sizes[1] / 2;
  const int EP = E + N_NODES;

  float* ws = (float*)d_ws;
  size_t off = 0;
  float* xl      = ws + off; off += (size_t)N_NODES * HEADS * HID;
  float* hA      = ws + off; off += (size_t)N_NODES * HID;
  float* hB      = ws + off; off += (size_t)N_NODES * HID;
  float* alpha_s = ws + off; off += (size_t)N_NODES * HEADS;
  float* alpha_d = ws + off; off += (size_t)N_NODES * HEADS;
  unsigned* emax = (unsigned*)(ws + off); off += (size_t)N_NODES * HEADS;
  float* denom   = ws + off; off += (size_t)N_NODES * HEADS;
  float* ee      = ws + off; off += (size_t)EP * HEADS;
  float* acc     = ws + off; off += (size_t)N_NODES * HID;
  float* hidb    = ws + off; off += (size_t)N_NODES * (HID / 2);

  const int Mblocks = (N_NODES + 127) / 128;  // 79

  // prelude: h = elu(x @ Wp + bp)   ([10000,512] @ [512,128])
  wmma_gemm_kernel<<<Mblocks * (HID / 64), 256, 0, stream>>>(
      x, Wp, bp, hA, N_NODES, HID, IN_DIM, 1);

  float* hcur = hA;
  float* hnext = hB;
  for (int l = 0; l < LAYERS; ++l) {
    // xl = h @ Wl[l]   ([10000,128] @ [128,1024])
    wmma_gemm_kernel<<<Mblocks * ((HEADS * HID) / 64), 256, 0, stream>>>(
        hcur, Wl + (size_t)l * HID * HEADS * HID, nullptr, xl, N_NODES, HEADS * HID, HID, 0);
    alpha_kernel<<<(N_NODES * HEADS + 255) / 256, 256, 0, stream>>>(
        xl, att_src + (size_t)l * HEADS * HID, att_dst + (size_t)l * HEADS * HID,
        alpha_s, alpha_d);
    zero_kernel<<<(N_NODES * HEADS + 255) / 256, 256, 0, stream>>>((float*)emax, N_NODES * HEADS);
    zero_kernel<<<(N_NODES * HEADS + 255) / 256, 256, 0, stream>>>(denom, N_NODES * HEADS);
    zero_kernel<<<(N_NODES * HID + 255) / 256, 256, 0, stream>>>(acc, N_NODES * HID);
    edge_max_kernel<<<(EP * HEADS + 255) / 256, 256, 0, stream>>>(ei, E, EP, alpha_s, alpha_d, emax);
    edge_exp_kernel<<<(EP * HEADS + 255) / 256, 256, 0, stream>>>(ei, E, EP, alpha_s, alpha_d,
                                                                  emax, ee, denom);
    message_kernel<<<EP, HID, 0, stream>>>(ei, E, xl, ee, denom, acc);
    post_kernel<<<(N_NODES * HID + 255) / 256, 256, 0, stream>>>(
        acc, hcur, bl + (size_t)l * HID, gamma + (size_t)l * HID, beta + (size_t)l * HID, hnext);
    float* t = hcur; hcur = hnext; hnext = t;
  }

  // hid = elu(h @ W1 + b1)   ([10000,128] @ [128,64])
  wmma_gemm_kernel<<<Mblocks * ((HID / 2) / 64), 256, 0, stream>>>(
      hcur, W1, b1, hidb, N_NODES, HID / 2, HID, 1);
  // out = hid @ W2 + b2
  out_kernel<<<(N_NODES * NCLS + 255) / 256, 256, 0, stream>>>(hidb, W2, b2, (float*)d_out);
}
